// Net_43061342110447
// MI455X (gfx1250) — matrix operand order repaired and checked by reading, not compiled, not run on hardware
//
#include <hip/hip_runtime.h>
#include <hip/hip_bf16.h>
#include <math.h>

typedef __attribute__((ext_vector_type(16))) __bf16 v16bf;
typedef __attribute__((ext_vector_type(8)))  __bf16 v8bf;
typedef __attribute__((ext_vector_type(8)))  float  v8f;

static __device__ __forceinline__ float sigmoidf_(float x) { return 1.0f / (1.0f + expf(-x)); }

// ---------------------------------------------------------------- utility kernels

__global__ void fill_f32(float* __restrict__ p, float v, int n) {
    int i = blockIdx.x * blockDim.x + threadIdx.x;
    if (i < n) p[i] = v;
}

__global__ void deg_accum(const long long* __restrict__ col, float* __restrict__ deg, int E) {
    int i = blockIdx.x * blockDim.x + threadIdx.x;
    if (i < E) atomicAdd(&deg[(int)col[i]], 1.0f);
}

__global__ void rsqrt_inplace(float* __restrict__ p, int n) {
    int i = blockIdx.x * blockDim.x + threadIdx.x;
    if (i < n) p[i] = rsqrtf(p[i]);   // deg >= 1 (self loop) so no zero guard needed
}

__global__ void f32_to_bf16(const float* __restrict__ in, __bf16* __restrict__ out, int n, int relu) {
    int i = blockIdx.x * blockDim.x + threadIdx.x;
    if (i < n) {
        float v = in[i];
        if (relu) v = fmaxf(v, 0.0f);
        out[i] = (__bf16)v;
    }
}

// W: K x M row-major (stored [in,out]) -> Wt: M x K row-major bf16 (K contiguous)
__global__ void transpose_w_bf16(const float* __restrict__ W, __bf16* __restrict__ Wt, int K, int M) {
    int i = blockIdx.x * blockDim.x + threadIdx.x;
    if (i < K * M) {
        int k = i / M, m = i % M;
        Wt[(size_t)m * K + k] = (__bf16)W[i];
    }
}

// ---------------------------------------------------------------- WMMA GEMM (bf16 in, f32 acc)
// C[Nn x M] = A[Nn x K] * B[K x M], with B supplied transposed as Bt[M x K].
// One wave computes a 16 x (16*G) strip: one A fragment per K-step reused across
// G B fragments / G accumulators. All fragment loads for a K-step are issued
// before any WMMA so the scheduler can clause them and overlap with the MMAs.

static __device__ __forceinline__ v16bf load_2x8(const __bf16* p0, const __bf16* p1) {
    v8bf lo = *reinterpret_cast<const v8bf*>(p0);
    v8bf hi = *reinterpret_cast<const v8bf*>(p1);
    return __builtin_shufflevector(lo, hi, 0, 1, 2, 3, 4, 5, 6, 7, 8, 9, 10, 11, 12, 13, 14, 15);
}

template <int G>
__global__ void __launch_bounds__(256) gemm_bf16_wmma(
    const __bf16* __restrict__ A,   // Nn x K row-major
    const __bf16* __restrict__ Bt,  // M x K row-major
    float* __restrict__ C,          // Nn x M row-major
    int Nn, int K, int M)
{
    const int lane    = threadIdx.x & 31;
    const int wave    = (blockIdx.x << 3) | (threadIdx.x >> 5);
    const int groupsM = (M >> 4) / G;              // column-tile groups per row strip
    const int tilesN  = Nn >> 4;
    if (wave >= groupsM * tilesN) return;          // wave-uniform: EXEC all-ones for WMMA

    const int tn = wave / groupsM;                 // output row tile
    const int tg = wave % groupsM;                 // output column group
    const int half = lane >> 4;
    const int l16  = lane & 15;

    // A fragment: lane holds row (tn*16 + l16); K pieces [half*8, half*8+8) and +16.
    const __bf16* arow = A + (size_t)(tn * 16 + l16) * K;
    // B fragments: lane holds column ((tg*G+g)*16 + l16); K pieces [half*8, half*8+8) and +16.
    const __bf16* brow[G];
#pragma unroll
    for (int g = 0; g < G; ++g)
        brow[g] = Bt + (size_t)((tg * G + g) * 16 + l16) * K;

    v8f acc[G] = {};
    for (int k0 = 0; k0 < K; k0 += 32) {
        // issue every load of this K-step first, then consume
        v16bf av = load_2x8(arow + k0 + half * 8, arow + k0 + 16 + half * 8);
        v16bf bv[G];
#pragma unroll
        for (int g = 0; g < G; ++g)
            bv[g] = load_2x8(brow[g] + k0 + half * 8, brow[g] + k0 + 16 + half * 8);
#pragma unroll
        for (int g = 0; g < G; ++g)
            acc[g] = __builtin_amdgcn_wmma_f32_16x16x32_bf16(false, av, false, bv[g],
                                                             (short)0, acc[g], false, false);
    }

    // C/D layout: VGPR v -> row v + 8*half, col l16.
    float* outbase = C + (size_t)(tn * 16) * M + tg * G * 16;
#pragma unroll
    for (int g = 0; g < G; ++g)
#pragma unroll
        for (int v = 0; v < 8; ++v)
            outbase[(size_t)(v + 8 * half) * M + g * 16 + l16] = acc[g][v];
}

// ---------------------------------------------------------------- GCN aggregation
// H[n][f] = dis[n]^2 * T[n][f] + b[f]   (self-loop term + bias)
__global__ void init_out(const float* __restrict__ T, const float* __restrict__ dis,
                         const float* __restrict__ b, float* __restrict__ H, int M, int total) {
    int i = blockIdx.x * blockDim.x + threadIdx.x;
    if (i < total) {
        int n = i / M;
        int f = i - n * M;
        float d = dis[n];
        H[i] = d * d * T[i] + b[f];
    }
}

// H[col[e]] += dis[row]*dis[col] * T[row[e]]   (one block per edge)
__global__ void scatter_edges(const float* __restrict__ T, const long long* __restrict__ row,
                              const long long* __restrict__ col, const float* __restrict__ dis,
                              float* __restrict__ H, int M) {
    int e = blockIdx.x;
    int r = (int)row[e], c = (int)col[e];
    float w = dis[r] * dis[c];
    const float* src = T + (size_t)r * M;
    float*       dst = H + (size_t)c * M;
    for (int f = threadIdx.x; f < M; f += blockDim.x)
        atomicAdd(&dst[f], w * src[f]);
}

// ---------------------------------------------------------------- Set2Set (q_star = h = 0 initially,
// so the Wih/Whh matmuls vanish and gates = bih + bhh exactly)
__global__ void set2set_q(const float* __restrict__ bih, const float* __restrict__ bhh,
                          float* __restrict__ q) {
    int j = threadIdx.x;
    if (j < 32) {
        float ig = bih[j]      + bhh[j];
        float gg = bih[64 + j] + bhh[64 + j];
        float og = bih[96 + j] + bhh[96 + j];
        float cs = sigmoidf_(ig) * tanhf(gg);   // sigmoid(f)*c0 term is zero (c0 = 0)
        q[j] = sigmoidf_(og) * tanhf(cs);
    }
}

// e[n] = h[n] . q ; per-block partial max
__global__ void attn_logits(const float* __restrict__ h, const float* __restrict__ q,
                            float* __restrict__ e, float* __restrict__ partmax, int Nn) {
    __shared__ float sm[256];
    int i = blockIdx.x * blockDim.x + threadIdx.x;
    float v = -INFINITY;
    if (i < Nn) {
        const float* hp = h + (size_t)i * 32;
        float s = 0.0f;
#pragma unroll
        for (int f = 0; f < 32; ++f) s += hp[f] * q[f];
        e[i] = s;
        v = s;
    }
    sm[threadIdx.x] = v;
    __syncthreads();
    for (int o = 128; o; o >>= 1) {
        if (threadIdx.x < o) sm[threadIdx.x] = fmaxf(sm[threadIdx.x], sm[threadIdx.x + o]);
        __syncthreads();
    }
    if (threadIdx.x == 0) partmax[blockIdx.x] = sm[0];
}

__global__ void reduce_max_k(const float* __restrict__ part, int n, float* __restrict__ maxv) {
    __shared__ float sm[256];
    float v = -INFINITY;
    for (int i = threadIdx.x; i < n; i += 256) v = fmaxf(v, part[i]);
    sm[threadIdx.x] = v;
    __syncthreads();
    for (int o = 128; o; o >>= 1) {
        if (threadIdx.x < o) sm[threadIdx.x] = fmaxf(sm[threadIdx.x], sm[threadIdx.x + o]);
        __syncthreads();
    }
    if (threadIdx.x == 0) maxv[0] = sm[0];
}

// sumexp += sum exp(e - max); r[f] += sum exp(e - max) * h[n][f]
__global__ void attn_accum(const float* __restrict__ h, const float* __restrict__ e,
                           const float* __restrict__ maxv, float* __restrict__ sumexp,
                           float* __restrict__ r, int Nn) {
    const float m = maxv[0];
    float sloc = 0.0f;
    float rloc[32];
#pragma unroll
    for (int f = 0; f < 32; ++f) rloc[f] = 0.0f;

    for (int i = blockIdx.x * blockDim.x + threadIdx.x; i < Nn; i += gridDim.x * blockDim.x) {
        float p = expf(e[i] - m);
        sloc += p;
        const float* hp = h + (size_t)i * 32;
#pragma unroll
        for (int f = 0; f < 32; ++f) rloc[f] += p * hp[f];
    }

    __shared__ float sm[256];
    int t = threadIdx.x;
    sm[t] = sloc;
    __syncthreads();
    for (int o = 128; o; o >>= 1) {
        if (t < o) sm[t] += sm[t + o];
        __syncthreads();
    }
    if (t == 0) atomicAdd(sumexp, sm[0]);
    for (int f = 0; f < 32; ++f) {
        __syncthreads();
        sm[t] = rloc[f];
        __syncthreads();
        for (int o = 128; o; o >>= 1) {
            if (t < o) sm[t] += sm[t + o];
            __syncthreads();
        }
        if (t == 0) atomicAdd(&r[f], sm[0]);
    }
}

// q_star = [q, r/sumexp] -> 64 -> relu(256) -> 128 -> 32 -> 1
__global__ void mlp_head(const float* __restrict__ q, const float* __restrict__ r,
                         const float* __restrict__ sumexp,
                         const float* __restrict__ Wl1, const float* __restrict__ bl1,
                         const float* __restrict__ Wl2, const float* __restrict__ bl2,
                         const float* __restrict__ Wl3, const float* __restrict__ bl3,
                         const float* __restrict__ Wl4, const float* __restrict__ bl4,
                         float* __restrict__ out) {
    __shared__ float qs[64], o1[256], o2[128], o3[32];
    int t = threadIdx.x;
    if (t < 32) qs[t] = q[t];
    else if (t < 64) qs[t] = r[t - 32] / sumexp[0];
    __syncthreads();
    {
        float s = bl1[t];
        for (int k = 0; k < 64; ++k) s += qs[k] * Wl1[k * 256 + t];
        o1[t] = fmaxf(s, 0.0f);
    }
    __syncthreads();
    if (t < 128) {
        float s = bl2[t];
        for (int k = 0; k < 256; ++k) s += o1[k] * Wl2[k * 128 + t];
        o2[t] = s;
    }
    __syncthreads();
    if (t < 32) {
        float s = bl3[t];
        for (int k = 0; k < 128; ++k) s += o2[k] * Wl3[k * 32 + t];
        o3[t] = s;
    }
    __syncthreads();
    if (t == 0) {
        float s = bl4[0];
        for (int k = 0; k < 32; ++k) s += o3[k] * Wl4[k];
        out[0] = s;
    }
}

// ---------------------------------------------------------------- host side

extern "C" void kernel_launch(void* const* d_in, const int* in_sizes, int n_in,
                              void* d_out, int out_size, void* d_ws, size_t ws_size,
                              hipStream_t stream) {
    (void)n_in; (void)out_size; (void)ws_size;

    const float*     x   = (const float*)d_in[0];
    const long long* ei  = (const long long*)d_in[1];
    const float* W1 = (const float*)d_in[2];  const float* b1 = (const float*)d_in[3];
    const float* W2 = (const float*)d_in[4];  const float* b2 = (const float*)d_in[5];
    const float* W3 = (const float*)d_in[6];  const float* b3 = (const float*)d_in[7];
    // d_in[8] (Wih), d_in[9] (Whh) multiply zero vectors in the reference -> unused
    const float* bih = (const float*)d_in[10]; const float* bhh = (const float*)d_in[11];
    const float* Wl1 = (const float*)d_in[12]; const float* bl1 = (const float*)d_in[13];
    const float* Wl2 = (const float*)d_in[14]; const float* bl2 = (const float*)d_in[15];
    const float* Wl3 = (const float*)d_in[16]; const float* bl3 = (const float*)d_in[17];
    const float* Wl4 = (const float*)d_in[18]; const float* bl4 = (const float*)d_in[19];

    const int Nn = in_sizes[0] / 64;
    const int E  = in_sizes[1] / 2;
    const long long* erow = ei;
    const long long* ecol = ei + E;

    // workspace carve-up (256B aligned slabs)
    char* ws = (char*)d_ws;
    size_t off = 0;
    auto alloc = [&](size_t bytes) -> void* {
        void* p = ws + off;
        off = (off + bytes + 255) & ~(size_t)255;
        return p;
    };
    float*  dis    = (float*)alloc((size_t)Nn * 4);
    float*  bufA   = (float*)alloc((size_t)Nn * 256 * 4);   // GEMM output (pre-aggregation)
    float*  bufB   = (float*)alloc((size_t)Nn * 256 * 4);   // aggregated output
    __bf16* bufC   = (__bf16*)alloc((size_t)Nn * 256 * 2);  // bf16 GEMM input
    __bf16* W1t    = (__bf16*)alloc((size_t)64 * 256 * 2);
    __bf16* W2t    = (__bf16*)alloc((size_t)256 * 128 * 2);
    __bf16* W3t    = (__bf16*)alloc((size_t)128 * 32 * 2);
    float*  ebuf   = (float*)alloc((size_t)Nn * 4);
    float*  partmx = (float*)alloc((size_t)4096 * 4);
    float*  smallb = (float*)alloc((size_t)128 * 4);
    float* qv     = smallb;        // 32
    float* rv     = smallb + 32;   // 32
    float* sumexp = smallb + 64;   // 1
    float* maxv   = smallb + 65;   // 1

    auto cdiv = [](int a, int b) { return (a + b - 1) / b; };

    // ---- degree -> dis = rsqrt(deg), deg includes self-loop
    fill_f32<<<cdiv(Nn, 256), 256, 0, stream>>>(dis, 1.0f, Nn);
    deg_accum<<<cdiv(E, 256), 256, 0, stream>>>(ecol, dis, E);
    rsqrt_inplace<<<cdiv(Nn, 256), 256, 0, stream>>>(dis, Nn);

    // ---- precision prep
    f32_to_bf16<<<cdiv(Nn * 64, 256), 256, 0, stream>>>(x, bufC, Nn * 64, 0);
    transpose_w_bf16<<<cdiv(64 * 256, 256), 256, 0, stream>>>(W1, W1t, 64, 256);
    transpose_w_bf16<<<cdiv(256 * 128, 256), 256, 0, stream>>>(W2, W2t, 256, 128);
    transpose_w_bf16<<<cdiv(128 * 32, 256), 256, 0, stream>>>(W3, W3t, 128, 32);

    const int tilesN = Nn / 16;

    // ---- layer 1: K=64 -> M=256, relu   (G=4: wave computes 16x64 strip)
    {
        const int M = 256, K = 64, G = 4;
        int waves = tilesN * ((M / 16) / G);
        gemm_bf16_wmma<G><<<cdiv(waves, 8), 256, 0, stream>>>(bufC, W1t, bufA, Nn, K, M);
        init_out<<<cdiv(Nn * M, 256), 256, 0, stream>>>(bufA, dis, b1, bufB, M, Nn * M);
        scatter_edges<<<E, M, 0, stream>>>(bufA, erow, ecol, dis, bufB, M);
        f32_to_bf16<<<cdiv(Nn * M, 256), 256, 0, stream>>>(bufB, bufC, Nn * M, 1);
    }
    // ---- layer 2: K=256 -> M=128   (G=4)
    {
        const int M = 128, K = 256, G = 4;
        int waves = tilesN * ((M / 16) / G);
        gemm_bf16_wmma<G><<<cdiv(waves, 8), 256, 0, stream>>>(bufC, W2t, bufA, Nn, K, M);
        init_out<<<cdiv(Nn * M, 256), 256, 0, stream>>>(bufA, dis, b2, bufB, M, Nn * M);
        scatter_edges<<<E, M, 0, stream>>>(bufA, erow, ecol, dis, bufB, M);
        f32_to_bf16<<<cdiv(Nn * M, 256), 256, 0, stream>>>(bufB, bufC, Nn * M, 0);
    }
    // ---- layer 3: K=128 -> M=32 ; keep f32 result in bufB (= h)   (G=2)
    {
        const int M = 32, K = 128, G = 2;
        int waves = tilesN * ((M / 16) / G);
        gemm_bf16_wmma<G><<<cdiv(waves, 8), 256, 0, stream>>>(bufC, W3t, bufA, Nn, K, M);
        init_out<<<cdiv(Nn * M, 256), 256, 0, stream>>>(bufA, dis, b3, bufB, M, Nn * M);
        scatter_edges<<<E, M, 0, stream>>>(bufA, erow, ecol, dis, bufB, M);
    }

    // ---- Set2Set + attention + head
    set2set_q<<<1, 32, 0, stream>>>(bih, bhh, qv);
    int nb = cdiv(Nn, 256);
    attn_logits<<<nb, 256, 0, stream>>>(bufB, qv, ebuf, partmx, Nn);
    reduce_max_k<<<1, 256, 0, stream>>>(partmx, nb, maxv);
    fill_f32<<<1, 64, 0, stream>>>(rv, 0.0f, 33);   // zero r[32] and sumexp (adjacent)
    int nb2 = nb < 512 ? nb : 512;
    attn_accum<<<nb2, 256, 0, stream>>>(bufB, ebuf, maxv, sumexp, rv, Nn);
    mlp_head<<<1, 256, 0, stream>>>(qv, rv, sumexp, Wl1, bl1, Wl2, bl2, Wl3, bl3, Wl4, bl4,
                                    (float*)d_out);
}
